// Attention_35235911696595
// MI455X (gfx1250) — matrix operand order repaired and checked by reading, not compiled, not run on hardware
//
#include <hip/hip_runtime.h>
#include <hip/hip_bf16.h>

// ---------------- problem constants ----------------
constexpr int B_   = 2;
constexpr int N_   = 1024;
constexpr int DIM_ = 2048;
constexpr int H_   = 16;
constexpr int DH_  = 128;
constexpr int MAXPOS_ = 16;
constexpr float QK_SCALE_ = 10.0f;
constexpr float NEGV_ = -1e30f;

// ---------------- workspace layout ----------------
constexpr size_t SZ_QKV = (size_t)B_ * N_ * H_ * DH_;   // 4,194,304
constexpr size_t SZ_SIM = (size_t)B_ * H_ * N_ * N_;    // 33,554,432
constexpr size_t OFF_Q  = 0;
constexpr size_t OFF_K  = OFF_Q  + SZ_QKV;
constexpr size_t OFF_V  = OFF_K  + SZ_QKV;
constexpr size_t OFF_VG = OFF_V  + SZ_QKV;
constexpr size_t OFF_HG = OFF_VG + SZ_QKV;              // B*N*H = 32768
constexpr size_t OFF_SA = OFF_HG + (size_t)B_ * N_ * H_;
constexpr size_t OFF_SB = OFF_SA + SZ_SIM;
constexpr size_t OFF_H16= OFF_SB + SZ_SIM;              // f16 pool (16B aligned)
constexpr size_t HSZ_MAT = (size_t)DIM_ * H_ * DH_;     // 4,194,304
constexpr size_t H_X   = 0;
constexpr size_t H_WQ  = H_X   + HSZ_MAT;
constexpr size_t H_WK  = H_WQ  + HSZ_MAT;
constexpr size_t H_WV  = H_WK  + HSZ_MAT;
constexpr size_t H_WVG = H_WV  + HSZ_MAT;
constexpr size_t H_WO  = H_WVG + HSZ_MAT;
constexpr size_t H_WHG = H_WO  + HSZ_MAT;               // 16*2048
constexpr size_t H_Q16 = H_WHG + (size_t)H_ * DIM_;
constexpr size_t H_K16 = H_Q16 + SZ_QKV;
constexpr size_t H_VT  = H_K16 + SZ_QKV;
constexpr size_t H_A2  = H_VT  + SZ_QKV;
constexpr size_t H_OP  = H_A2  + SZ_SIM;

// ---------------- WMMA plumbing (CDNA5 wave32) ----------------
typedef __attribute__((ext_vector_type(16))) _Float16 v16h;
typedef __attribute__((ext_vector_type(8)))  _Float16 v8h;
typedef __attribute__((ext_vector_type(8)))  float    v8f;

__device__ __forceinline__ v8f wmma_f16(v16h a, v16h b, v8f c) {
  return __builtin_amdgcn_wmma_f32_16x16x32_f16(
      false, a, false, b, (short)0, c, false, false);
}
__device__ __forceinline__ v16h cat8(v8h lo, v8h hi) {
  return __builtin_shufflevector(lo, hi, 0,1,2,3,4,5,6,7,8,9,10,11,12,13,14,15);
}
// A frag: ap = &A16[(m0+lane%16)*lda + k0 + (lane/16)*16]; two b128 loads
__device__ __forceinline__ v16h load_a_h(const _Float16* ap) {
  return cat8(*(const v8h*)ap, *(const v8h*)(ap + 8));
}
// B frag: bp = &Bt16[(n0+lane%16)*ldb + k0 + (lane/16)*16]; two b128 loads
__device__ __forceinline__ v16h load_b_h(const _Float16* bp) {
  return cat8(*(const v8h*)bp, *(const v8h*)(bp + 8));
}

// ---------------- f32 -> f16 convert ----------------
__global__ void convert_h_kernel(const float* __restrict__ src, _Float16* __restrict__ dst,
                                 int n) {
  int idx = (blockIdx.x * 256 + threadIdx.x) * 4;
  if (idx + 3 < n) {
    float4 v = *(const float4*)(src + idx);
    dst[idx + 0] = (_Float16)v.x; dst[idx + 1] = (_Float16)v.y;
    dst[idx + 2] = (_Float16)v.z; dst[idx + 3] = (_Float16)v.w;
  } else {
    for (int q = 0; q < 4 && idx + q < n; ++q) dst[idx + q] = (_Float16)src[idx + q];
  }
}

// ---------------- tiled transpose + f16 convert: dst[c][r] = src[r][c] --------
__global__ void transpose_h_kernel(const float* __restrict__ src, _Float16* __restrict__ dst,
                                   int R, int Ccols) {
  __shared__ _Float16 tile[32][33];
  const int c0 = blockIdx.x * 32, r0 = blockIdx.y * 32;
  const int tx = threadIdx.x & 31, ty = threadIdx.x >> 5;  // 32 x 8
  for (int a = ty; a < 32; a += 8) {
    int r = r0 + a, c = c0 + tx;
    tile[a][tx] = (r < R && c < Ccols) ? (_Float16)src[(size_t)r * Ccols + c] : (_Float16)0.f;
  }
  __syncthreads();
  for (int a = ty; a < 32; a += 8) {
    int c = c0 + a, r = r0 + tx;
    if (c < Ccols && r < R) dst[(size_t)c * R + r] = tile[tx][a];
  }
}

// ---------------- per-(b,h) V transpose: Vt[bh][d][j] = V[b][j][h][d] ----------
__global__ void v_transpose_kernel(const float* __restrict__ V, _Float16* __restrict__ Vt) {
  __shared__ _Float16 tile[32][33];
  const int z = blockIdx.z, b = z >> 4, h = z & 15;
  const float* src = V + ((size_t)b * N_ * H_ + h) * DH_;
  _Float16* dst = Vt + (size_t)z * DH_ * N_;
  const int j0 = blockIdx.y * 32, d0 = blockIdx.x * 32;
  const int tx = threadIdx.x & 31, ty = threadIdx.x >> 5;
  for (int a = ty; a < 32; a += 8)
    tile[a][tx] = (_Float16)src[(size_t)(j0 + a) * (H_ * DH_) + d0 + tx];
  __syncthreads();
  for (int a = ty; a < 32; a += 8)
    dst[(size_t)(d0 + a) * N_ + j0 + tx] = tile[tx][a];
}

// ---------------- f16 GEMM, full tiles only (M%256==0, N%64==0) ----------------
// 8 waves; wave owns 32(M) x 64(N): 8 accumulators, branch-free k-loop.
__global__ void __launch_bounds__(256, 1)
gemm_h_kernel(const _Float16* __restrict__ A, int lda,
              const _Float16* __restrict__ Bt, int ldb,
              float* __restrict__ C, int ldc, int Kk) {
  const int wave = threadIdx.x >> 5;
  const unsigned lane = threadIdx.x & 31u;
  const int m0 = blockIdx.y * 256 + wave * 32;
  const int n0 = blockIdx.x * 64;
  const int lm = lane & 15, lh = lane >> 4;
  const _Float16* ap0 = A + (size_t)(m0 + lm) * lda + lh * 16;
  const _Float16* ap1 = ap0 + (size_t)16 * lda;
  const _Float16* bp0 = Bt + (size_t)(n0 + lm) * ldb + lh * 16;
  v8f z = {0.f,0.f,0.f,0.f,0.f,0.f,0.f,0.f};
  v8f acc0[4] = {z, z, z, z};
  v8f acc1[4] = {z, z, z, z};
#pragma unroll 1
  for (int k = 0; k < Kk; k += 32) {
    v16h a0 = load_a_h(ap0 + k);
    v16h a1 = load_a_h(ap1 + k);
#pragma unroll
    for (int t = 0; t < 4; ++t) {
      v16h b = load_b_h(bp0 + (size_t)t * 16 * ldb + k);
      acc0[t] = wmma_f16(a0, b, acc0[t]);
      acc1[t] = wmma_f16(a1, b, acc1[t]);
    }
  }
  const int nl = lm, mh = lh * 8;
#pragma unroll
  for (int t = 0; t < 4; ++t) {
    int n = n0 + t * 16 + nl;
#pragma unroll
    for (int v = 0; v < 8; ++v) {
      C[(size_t)(m0 + mh + v) * ldc + n]      = acc0[t][v];
      C[(size_t)(m0 + 16 + mh + v) * ldc + n] = acc1[t][v];
    }
  }
}

// ---------------- f16 GEMM, N==16 (head-gate projection), guard-free ------------
__global__ void __launch_bounds__(256, 1)
gemm_h_n16_kernel(const _Float16* __restrict__ A, int lda,
                  const _Float16* __restrict__ Bt, int ldb,
                  float* __restrict__ C, int ldc, int Kk) {
  const int wave = threadIdx.x >> 5;
  const unsigned lane = threadIdx.x & 31u;
  const int m0 = blockIdx.y * 256 + wave * 32;
  const int lm = lane & 15, lh = lane >> 4;
  const _Float16* ap0 = A + (size_t)(m0 + lm) * lda + lh * 16;
  const _Float16* ap1 = ap0 + (size_t)16 * lda;
  const _Float16* bp0 = Bt + (size_t)lm * ldb + lh * 16;
  v8f acc0 = {0.f,0.f,0.f,0.f,0.f,0.f,0.f,0.f};
  v8f acc1 = acc0;
#pragma unroll 1
  for (int k = 0; k < Kk; k += 32) {
    v16h b = load_b_h(bp0 + k);
    acc0 = wmma_f16(load_a_h(ap0 + k), b, acc0);
    acc1 = wmma_f16(load_a_h(ap1 + k), b, acc1);
  }
  const int nl = lm, mh = lh * 8;
#pragma unroll
  for (int v = 0; v < 8; ++v) {
    C[(size_t)(m0 + mh + v) * ldc + nl]      = acc0[v];
    C[(size_t)(m0 + 16 + mh + v) * ldc + nl] = acc1[v];
  }
}

// ---------------- fused l2-norm + rotary; writes f32 in place + f16 copy --------
__global__ void qk_prep_kernel(float* __restrict__ Q, float* __restrict__ Kt,
                               _Float16* __restrict__ Q16, _Float16* __restrict__ K16,
                               const float* __restrict__ freqs) {
  float* Tp = (blockIdx.y == 0) ? Q : Kt;
  _Float16* Hp = (blockIdx.y == 0) ? Q16 : K16;
  const int r = blockIdx.x;                    // (b*N + n)*H + h
  const int n = (r / H_) % N_;
  const int d = threadIdx.x;                   // 128 threads
  float* row = Tp + (size_t)r * DH_;
  __shared__ float sh[DH_];
  __shared__ float ssum[DH_];
  float v = row[d];
  ssum[d] = v * v;
  __syncthreads();
  for (int off = 64; off > 0; off >>= 1) {
    if (d < off) ssum[d] += ssum[d + off];
    __syncthreads();
  }
  float nrm = sqrtf(ssum[0]);
  float nv = v / fmaxf(nrm, 1e-12f);
  sh[d] = nv;
  __syncthreads();
  float rot = (d & 1) ? sh[d - 1] : -sh[d + 1];
  float f = freqs[(size_t)n * DH_ + d];
  float o = nv * cosf(f) + rot * sinf(f);
  row[d] = o;
  Hp[(size_t)r * DH_ + d] = (_Float16)o;
}

// ---------------- QK^T per 16x16 tile (lower-triangular tiles only) -------------
__global__ void __launch_bounds__(256, 1)
sim_qk_kernel(const _Float16* __restrict__ Q16, const _Float16* __restrict__ K16,
              float* __restrict__ simraw) {
  const int wave = threadIdx.x >> 5;
  const int gid = blockIdx.x * 8 + wave;
  const int TPH = (N_ / 16) * (N_ / 16);       // 4096
  const int bh = gid / TPH;
  if (bh >= B_ * H_) return;
  const int t = gid % TPH;
  const int it = t >> 6, jt = t & 63;
  if (jt > it) return;
  const int b = bh >> 4, h = bh & 15;
  const unsigned lane = threadIdx.x & 31u;
  const int lm = lane & 15, lh = lane >> 4;
  const _Float16* ap = Q16 + (((size_t)(b * N_ + it * 16 + lm) * H_ + h) * DH_) + lh * 16;
  const _Float16* bp = K16 + (((size_t)(b * N_ + jt * 16 + lm) * H_ + h) * DH_) + lh * 16;
  v8f acc = {0.f,0.f,0.f,0.f,0.f,0.f,0.f,0.f};
#pragma unroll
  for (int k = 0; k < DH_; k += 32)
    acc = wmma_f16(load_a_h(ap + k), load_b_h(bp + k), acc);
  const int nl = lm, mh = lh * 8;
  float* dst = simraw + ((size_t)bh * N_ + it * 16) * N_ + jt * 16;
#pragma unroll
  for (int v = 0; v < 8; ++v)
    dst[(size_t)(mh + v) * N_ + nl] = acc[v] * QK_SCALE_;
}

// ---------------- th_pre head-mix + causal mask -> f32 ----------------
__global__ void mix_pre_kernel(const float* __restrict__ src, float* __restrict__ dst,
                               const float* __restrict__ Mh) {
  size_t idx = (size_t)blockIdx.x * blockDim.x + threadIdx.x;
  const size_t total = (size_t)B_ * H_ * N_ * N_;
  if (idx >= total) return;
  const int j = (int)(idx % N_);
  size_t r = idx / N_;
  const int i = (int)(r % N_);
  const int h = (int)((r / N_) % H_);
  if (j > i) { dst[idx] = NEGV_; return; }
  const size_t base = idx - (size_t)h * N_ * N_;
  float s = 0.f;
#pragma unroll
  for (int g = 0; g < H_; ++g)
    s += Mh[h * H_ + g] * src[base + (size_t)g * N_ * N_];
  dst[idx] = s;
}

// ---------------- th_post head-mix -> f16 ----------------
__global__ void mix_post_kernel(const float* __restrict__ src, _Float16* __restrict__ dst,
                                const float* __restrict__ Mh) {
  size_t idx = (size_t)blockIdx.x * blockDim.x + threadIdx.x;
  const size_t total = (size_t)B_ * H_ * N_ * N_;
  if (idx >= total) return;
  const int h = (int)((idx / ((size_t)N_ * N_)) % H_);
  const size_t base = idx - (size_t)h * N_ * N_;
  float s = 0.f;
#pragma unroll
  for (int g = 0; g < H_; ++g)
    s += Mh[h * H_ + g] * src[base + (size_t)g * N_ * N_];
  dst[idx] = (_Float16)s;
}

// ---------------- CoPE + softmax, one row per block ----------------
__global__ void cope_softmax_kernel(float* __restrict__ sim, const float* __restrict__ Q,
                                    const float* __restrict__ cope) {
  constexpr int PER = N_ / 256;
  const int r = blockIdx.x;
  const int i  = r % N_;
  const int bh = r / N_;
  const int h = bh % H_, b = bh / H_;
  float* row = sim + ((size_t)bh * N_ + i) * N_;
  const float* qrow = Q + (((size_t)b * N_ + i) * H_ + h) * DH_;
  __shared__ float sscan[256];
  __shared__ float slog[MAXPOS_];
  __shared__ float sred[16][17];
  const int t = threadIdx.x;
  float s[PER], g[PER], e[PER];
  float localsum = 0.f;
#pragma unroll
  for (int q = 0; q < PER; ++q) {
    s[q] = row[t * PER + q];
    g[q] = 1.f / (1.f + expf(-s[q]));
    localsum += g[q];
  }
  sscan[t] = localsum;
  __syncthreads();
  for (int off = 1; off < 256; off <<= 1) {
    float vp = (t >= off) ? sscan[t - off] : 0.f;
    __syncthreads();
    sscan[t] += vp;
    __syncthreads();
  }
  const float total = sscan[255];
  const float excl = sscan[t] - localsum;
  {
    const int p = t >> 4, l = t & 15;
    float a = 0.f;
    for (int dd = l; dd < DH_; dd += 16) a += qrow[dd] * cope[p * DH_ + dd];
    sred[p][l] = a;
  }
  __syncthreads();
  if (t < MAXPOS_) {
    float a = 0.f;
#pragma unroll
    for (int l = 0; l < 16; ++l) a += sred[t][l];
    slog[t] = a;
  }
  __syncthreads();
  float run = excl;
  float mx = -3.0e38f;
#pragma unroll
  for (int q = 0; q < PER; ++q) {
    float pos = total - run;
    run += g[q];
    pos = fminf(fmaxf(pos, 0.f), (float)(MAXPOS_ - 1));
    float pf = floorf(pos);
    float w  = pos - pf;
    int ic = (int)ceilf(pos); if (ic > MAXPOS_ - 1) ic = MAXPOS_ - 1;
    int fl = (int)pf;
    s[q] += slog[ic] * w + slog[fl] * (1.f - w);
    mx = fmaxf(mx, s[q]);
  }
  __syncthreads();
  sscan[t] = mx;
  __syncthreads();
  for (int off = 128; off > 0; off >>= 1) {
    if (t < off) sscan[t] = fmaxf(sscan[t], sscan[t + off]);
    __syncthreads();
  }
  mx = sscan[0];
  __syncthreads();
  float esum = 0.f;
#pragma unroll
  for (int q = 0; q < PER; ++q) { e[q] = expf(s[q] - mx); esum += e[q]; }
  sscan[t] = esum;
  __syncthreads();
  for (int off = 128; off > 0; off >>= 1) {
    if (t < off) sscan[t] += sscan[t + off];
    __syncthreads();
  }
  const float inv = 1.f / sscan[0];
#pragma unroll
  for (int q = 0; q < PER; ++q) row[t * PER + q] = e[q] * inv;
}

// ---------------- attn @ V with fused gates; writes f16 OP ----------------
__global__ void __launch_bounds__(256, 1)
av_gemm_kernel(const _Float16* __restrict__ A2, const _Float16* __restrict__ Vt,
               const float* __restrict__ HGp, const float* __restrict__ VGp,
               const float* __restrict__ b_h, const float* __restrict__ b_v,
               _Float16* __restrict__ OutPre) {
  const int wave = threadIdx.x >> 5;
  const int gid = blockIdx.x * 8 + wave;
  const int bh = gid >> 7;
  if (bh >= B_ * H_) return;
  const int t = gid & 127;
  const int it = t >> 1, dt = t & 1;
  const int b = bh >> 4, h = bh & 15;
  const unsigned lane = threadIdx.x & 31u;
  const int lm = lane & 15, lh = lane >> 4;
  const _Float16* ap = A2 + ((size_t)bh * N_ + it * 16 + lm) * N_ + lh * 16;
  const _Float16* bp = Vt + ((size_t)bh * DH_ + dt * 64 + lm) * N_ + lh * 16;
  v8f z = {0.f,0.f,0.f,0.f,0.f,0.f,0.f,0.f};
  v8f acc[4] = {z, z, z, z};
  const int kmax = ((it * 16 + 16) + 31) & ~31;   // causal clip
#pragma unroll 1
  for (int k = 0; k < kmax; k += 32) {
    v16h a = load_a_h(ap + k);
#pragma unroll
    for (int tt = 0; tt < 4; ++tt) {
      v16h bb = load_b_h(bp + (size_t)tt * 16 * N_ + k);
      acc[tt] = wmma_f16(a, bb, acc[tt]);
    }
  }
  const int nl = lm, mh = lh * 8;
#pragma unroll
  for (int tt = 0; tt < 4; ++tt) {
    const int d0 = dt * 64 + tt * 16 + nl;
#pragma unroll
    for (int v = 0; v < 8; ++v) {
      const int i = it * 16 + mh + v;
      const size_t rowi = (size_t)b * N_ + i;
      float hg = 1.f / (1.f + expf(-(HGp[rowi * H_ + h] + b_h[h])));
      float vp = VGp[rowi * (H_ * DH_) + h * DH_ + d0] + b_v[h * DH_ + d0];
      float vg = 1.f / (1.f + expf(-vp));
      OutPre[rowi * (H_ * DH_) + h * DH_ + d0] = (_Float16)(acc[tt][v] * hg * vg);
    }
  }
}

// ---------------- launch ----------------
extern "C" void kernel_launch(void* const* d_in, const int* in_sizes, int n_in,
                              void* d_out, int out_size, void* d_ws, size_t ws_size,
                              hipStream_t stream) {
  const float* x        = (const float*)d_in[0];
  const float* freqs    = (const float*)d_in[1];
  const float* w_q      = (const float*)d_in[2];
  const float* w_k      = (const float*)d_in[3];
  const float* w_v      = (const float*)d_in[4];
  const float* cope_emb = (const float*)d_in[5];
  const float* th_pre   = (const float*)d_in[6];
  const float* th_post  = (const float*)d_in[7];
  const float* w_vgate  = (const float*)d_in[8];
  const float* b_vgate  = (const float*)d_in[9];
  const float* w_hgate  = (const float*)d_in[10];
  const float* b_hgate  = (const float*)d_in[11];
  const float* w_out    = (const float*)d_in[12];
  float* out = (float*)d_out;

  float* ws = (float*)d_ws;
  float* Q  = ws + OFF_Q;
  float* Kt = ws + OFF_K;
  float* V  = ws + OFF_V;
  float* VG = ws + OFF_VG;
  float* HG = ws + OFF_HG;
  float* SA = ws + OFF_SA;
  float* SB = ws + OFF_SB;
  _Float16* hp  = (_Float16*)(ws + OFF_H16);
  _Float16* X16 = hp + H_X;
  _Float16* WQ  = hp + H_WQ;
  _Float16* WK  = hp + H_WK;
  _Float16* WV  = hp + H_WV;
  _Float16* WVG = hp + H_WVG;
  _Float16* WO  = hp + H_WO;
  _Float16* WHG = hp + H_WHG;
  _Float16* Q16 = hp + H_Q16;
  _Float16* K16 = hp + H_K16;
  _Float16* VT  = hp + H_VT;
  _Float16* A2  = hp + H_A2;
  _Float16* OP  = hp + H_OP;

  const int M  = B_ * N_;    // 2048
  const int HD = H_ * DH_;   // 2048
  dim3 blk(256);

  // 0) operand staging
  convert_h_kernel<<<(unsigned)((HSZ_MAT / 4 + 255) / 256), blk, 0, stream>>>(x, X16, (int)HSZ_MAT);
  transpose_h_kernel<<<dim3(64, 64), blk, 0, stream>>>(w_q,     WQ,  DIM_, HD);
  transpose_h_kernel<<<dim3(64, 64), blk, 0, stream>>>(w_k,     WK,  DIM_, HD);
  transpose_h_kernel<<<dim3(64, 64), blk, 0, stream>>>(w_v,     WV,  DIM_, HD);
  transpose_h_kernel<<<dim3(64, 64), blk, 0, stream>>>(w_vgate, WVG, DIM_, HD);
  transpose_h_kernel<<<dim3(64, 64), blk, 0, stream>>>(w_out,   WO,  HD,  DIM_);
  transpose_h_kernel<<<dim3(1, 64),  blk, 0, stream>>>(w_hgate, WHG, DIM_, H_);

  // 1) projections
  dim3 gProj(HD / 64, M / 256);                 // (32, 8)
  gemm_h_kernel<<<gProj, blk, 0, stream>>>(X16, DIM_, WQ,  DIM_, Q,  HD, DIM_);
  gemm_h_kernel<<<gProj, blk, 0, stream>>>(X16, DIM_, WK,  DIM_, Kt, HD, DIM_);
  gemm_h_kernel<<<gProj, blk, 0, stream>>>(X16, DIM_, WV,  DIM_, V,  HD, DIM_);
  gemm_h_kernel<<<gProj, blk, 0, stream>>>(X16, DIM_, WVG, DIM_, VG, HD, DIM_);
  gemm_h_n16_kernel<<<dim3(1, M / 256), blk, 0, stream>>>(X16, DIM_, WHG, DIM_, HG, H_, DIM_);

  // 2) l2-norm + rotary; V transpose
  qk_prep_kernel<<<dim3(B_ * N_ * H_, 2), 128, 0, stream>>>(Q, Kt, Q16, K16, freqs);
  v_transpose_kernel<<<dim3(DH_ / 32, N_ / 32, B_ * H_), blk, 0, stream>>>(V, VT);

  // 3) sim = scale * Q K^T (causal-tile-pruned)
  sim_qk_kernel<<<B_ * H_ * (N_ / 16) * (N_ / 16) / 8, blk, 0, stream>>>(Q16, K16, SA);

  // 4) th_pre mix + causal mask -> SB
  {
    size_t total = (size_t)B_ * H_ * N_ * N_;
    mix_pre_kernel<<<(unsigned)((total + 255) / 256), blk, 0, stream>>>(SA, SB, th_pre);
  }

  // 5) CoPE + softmax in place on SB
  cope_softmax_kernel<<<B_ * H_ * N_, blk, 0, stream>>>(SB, Q, cope_emb);

  // 6) th_post mix -> A2 (f16)
  {
    size_t total = (size_t)B_ * H_ * N_ * N_;
    mix_post_kernel<<<(unsigned)((total + 255) / 256), blk, 0, stream>>>(SB, A2, th_post);
  }

  // 7) (attn @ V) * gates -> OP (f16)
  av_gemm_kernel<<<B_ * H_ * (N_ / 16) * (DH_ / 64) / 8, blk, 0, stream>>>(
      A2, VT, HG, VG, b_hgate, b_vgate, OP);

  // 8) final projection
  gemm_h_kernel<<<dim3(DIM_ / 64, M / 256), blk, 0, stream>>>(OP, HD, WO, HD, out, DIM_, HD);
}
